// SRUPos_68066641707708
// MI455X (gfx1250) — matrix operand order repaired and checked by reading, not compiled, not run on hardware
//
#include <hip/hip_runtime.h>
#include <hip/hip_bf16.h>

typedef __bf16 v16bf __attribute__((ext_vector_type(16)));
typedef float  v8f   __attribute__((ext_vector_type(8)));

union FragBF { unsigned u[8]; v16bf v; };

__device__ __forceinline__ unsigned short f2bf(float f) {
  unsigned u = __float_as_uint(f);
  u += 0x7fffu + ((u >> 16) & 1u);          // round-to-nearest-even
  return (unsigned short)(u >> 16);
}

__device__ __forceinline__ float sigmoidf_(float x) {
  return 1.0f / (1.0f + __expf(-x));
}

// ---------------------------------------------------------------------------
// Fused embedding gather + LayerNorm, writes X in (L,B,E) order. E=256 fixed.
// One wave32 per row; 8 elements per lane; shfl_xor wave reduction.
// ---------------------------------------------------------------------------
__global__ void embed_ln_kernel(const int* __restrict__ ids,
                                const float* __restrict__ tab,
                                const float* __restrict__ g,
                                const float* __restrict__ bvec,
                                float* __restrict__ X,
                                int Bn, int Ln) {
  const int E = 256;
  int lane = threadIdx.x & 31;
  int wave = threadIdx.x >> 5;
  int row = blockIdx.x * (blockDim.x >> 5) + wave;   // row = b*Ln + l
  if (row >= Bn * Ln) return;
  int b = row / Ln, l = row % Ln;
  const float* src = tab + (size_t)ids[row] * E;
  float v[8], s = 0.f, ss = 0.f;
#pragma unroll
  for (int j = 0; j < 8; ++j) {
    v[j] = src[lane + 32 * j];
    s += v[j]; ss += v[j] * v[j];
  }
#pragma unroll
  for (int off = 16; off > 0; off >>= 1) {
    s  += __shfl_xor(s, off);
    ss += __shfl_xor(ss, off);
  }
  float mean = s * (1.0f / E);
  float var  = ss * (1.0f / E) - mean * mean;
  float rstd = rsqrtf(var + 1e-5f);
  float* dst = X + ((size_t)l * Bn + b) * E;
#pragma unroll
  for (int j = 0; j < 8; ++j) {
    int c = lane + 32 * j;
    dst[c] = (v[j] - mean) * rstd * g[c] + bvec[c];
  }
}

// ---------------------------------------------------------------------------
// Per-row LayerNorm (256 cols) -> bf16 output (GEMM A operand).
// ---------------------------------------------------------------------------
__global__ void ln_rows_bf16_kernel(const float* __restrict__ X,
                                    const float* __restrict__ g,
                                    const float* __restrict__ bvec,
                                    unsigned short* __restrict__ Y,
                                    int rows) {
  const int E = 256;
  int lane = threadIdx.x & 31;
  int wave = threadIdx.x >> 5;
  int row = blockIdx.x * (blockDim.x >> 5) + wave;
  if (row >= rows) return;
  const float* src = X + (size_t)row * E;
  float v[8], s = 0.f, ss = 0.f;
#pragma unroll
  for (int j = 0; j < 8; ++j) {
    v[j] = src[lane + 32 * j];
    s += v[j]; ss += v[j] * v[j];
  }
#pragma unroll
  for (int off = 16; off > 0; off >>= 1) {
    s  += __shfl_xor(s, off);
    ss += __shfl_xor(ss, off);
  }
  float mean = s * (1.0f / E);
  float var  = ss * (1.0f / E) - mean * mean;
  float rstd = rsqrtf(var + 1e-5f);
  unsigned short* dst = Y + (size_t)row * E;
#pragma unroll
  for (int j = 0; j < 8; ++j) {
    int c = lane + 32 * j;
    dst[c] = f2bf((v[j] - mean) * rstd * g[c] + bvec[c]);
  }
}

// ---------------------------------------------------------------------------
// bf16 WMMA GEMM: C[M,N] = A[M,K] * Bt[N,K]^T (+bias). Each wave owns a 16x64
// tile (4x 16x16 f32 accumulators), K-loop step 32 using
// v_wmma_f32_16x16x32_bf16. Optional power-of-two (L,B)->(B,L) output-row
// remap via shifts: bshift = log2(B), lshift = log2(L); bshift < 0 disables.
// ---------------------------------------------------------------------------
__global__ void __launch_bounds__(128)
gemm_bf16_wmma(const unsigned short* __restrict__ A,
               const unsigned short* __restrict__ Bt,
               float* __restrict__ C,
               const float* __restrict__ bias,
               int M, int N, int K, int bshift, int lshift) {
  int lane = threadIdx.x & 31;
  int wave = threadIdx.x >> 5;
  int wid = blockIdx.x * (blockDim.x >> 5) + wave;
  int ntiles = N >> 6;                 // 64-wide N tiles
  int mtiles = M >> 4;
  if (wid >= mtiles * ntiles) return;  // wave-uniform -> EXEC stays all-1s
  int mt = wid / ntiles, nt = wid % ntiles;
  int m0 = mt << 4, n0 = nt << 6;
  int half = lane >> 4, lr = lane & 15;

  const unsigned short* arow = A + (size_t)(m0 + lr) * K;
  v8f zero = {0.f, 0.f, 0.f, 0.f, 0.f, 0.f, 0.f, 0.f};
  v8f acc[4];
#pragma unroll
  for (int t = 0; t < 4; ++t) acc[t] = zero;

  for (int k0 = 0; k0 < K; k0 += 32) {
    // A fragment (16x32 bf16): VGPR j<4: k = half*8 + 2j ; j>=4: 16 + half*8 + 2(j-4)
    FragBF a;
    int ka = k0 + half * 8;
#pragma unroll
    for (int j = 0; j < 4; ++j) a.u[j]     = *(const unsigned*)(arow + ka + 2 * j);
#pragma unroll
    for (int j = 0; j < 4; ++j) a.u[4 + j] = *(const unsigned*)(arow + ka + 16 + 2 * j);

    // B fragment (32x16 bf16): same K order as A: VGPR j<4: k = half*8 + 2j ; j>=4: 16 + half*8 + 2(j-4)
    int kb = k0 + half * 8;
#pragma unroll
    for (int t = 0; t < 4; ++t) {
      const unsigned short* brow = Bt + (size_t)(n0 + t * 16 + lr) * K;
      FragBF bfr;
#pragma unroll
      for (int j = 0; j < 8; ++j) bfr.u[j] = *(const unsigned*)(brow + kb + 2 * j + (j >= 4 ? 8 : 0));
      acc[t] = __builtin_amdgcn_wmma_f32_16x16x32_bf16(
          false, a.v, false, bfr.v, (short)0, acc[t], false, false);
    }
  }

  // C/D layout: VGPR j -> M = m0 + j + 8*half, N = n0 + t*16 + (lane&15)
  // Remap base (power-of-two shifts, hoisted out of store loop):
  //   row_out(m) = (m % B)*L + m/B ; m0 is 16-aligned so m%B = (m0%B) + (j+8h)
  size_t obase = (size_t)m0;
  size_t ostep = 1;
  if (bshift >= 0) {
    int mr0 = m0 & ((1 << bshift) - 1);
    obase = ((size_t)mr0 << lshift) + (size_t)(m0 >> bshift);
    ostep = (size_t)1 << lshift;
  }
#pragma unroll
  for (int t = 0; t < 4; ++t) {
    int n = n0 + t * 16 + lr;
    float bv = bias ? bias[n] : 0.f;
#pragma unroll
    for (int j = 0; j < 8; ++j) {
      size_t orow = obase + (size_t)(j + half * 8) * ostep;
      C[orow * N + n] = acc[t][j] + bv;
    }
  }
}

// ---------------------------------------------------------------------------
// SRU bidirectional recurrent scan. One thread per (b, dir, d) channel,
// sequential over L (fwd for dir=0, bwd for dir=1). U: (L,B,2,D,3) f32.
// Optionally also emits h as bf16 (Hbf) so the output projection can consume
// it directly without a separate conversion pass.
// ---------------------------------------------------------------------------
__global__ void sru_scan_kernel(const float* __restrict__ U,
                                const float* __restrict__ Xres,
                                const float* __restrict__ vc,
                                const float* __restrict__ bias,
                                float* __restrict__ Hout,
                                unsigned short* __restrict__ Hbf,
                                int Ln, int Bn, int Dd) {
  int idx = blockIdx.x * blockDim.x + threadIdx.x;
  int total = Bn * 2 * Dd;
  if (idx >= total) return;
  int d = idx % Dd;
  int dir = (idx / Dd) & 1;
  int b = idx / (2 * Dd);
  float vf = vc[dir * Dd + d];
  float vr = vc[2 * Dd + dir * Dd + d];
  float bf = bias[dir * Dd + d];
  float br = bias[2 * Dd + dir * Dd + d];
  size_t ustride = (size_t)Bn * 2 * Dd * 3;
  size_t xstride = (size_t)Bn * 2 * Dd;
  const float* ubase = U + ((size_t)b * 2 + dir) * Dd * 3 + (size_t)d * 3;
  size_t hoff = (size_t)b * 2 * Dd + (size_t)dir * Dd + d;
  const float* xbase = Xres + hoff;
  float*       hbase = Hout + hoff;
  unsigned short* hbfb = Hbf ? (Hbf + hoff) : nullptr;
  int l = dir ? (Ln - 1) : 0;
  int step = dir ? -1 : 1;
  float c = 0.f;
  for (int it = 0; it < Ln; ++it, l += step) {
    const float* up = ubase + (size_t)l * ustride;
    __builtin_prefetch(up + (long)step * (long)ustride, 0, 1);  // global_prefetch_b8
    float u0 = up[0], u1 = up[1], u2 = up[2];
    float x = xbase[(size_t)l * xstride];
    float f = sigmoidf_(u1 + vf * c + bf);
    c = f * c + (1.f - f) * u0;
    float r = sigmoidf_(u2 + vr * c + br);
    float cr = c > 0.f ? c : 0.f;
    float h = r * cr + (1.f - r) * x;
    hbase[(size_t)l * xstride] = h;
    if (hbfb) hbfb[(size_t)l * xstride] = f2bf(h);
  }
}

// ---------------------------------------------------------------------------
// Elementwise f32 -> bf16 and (K,N)f32 -> (N,K)bf16 transpose for weights.
// ---------------------------------------------------------------------------
__global__ void convert_bf16_kernel(const float* __restrict__ src,
                                    unsigned short* __restrict__ dst, size_t n) {
  size_t i = (size_t)blockIdx.x * blockDim.x + threadIdx.x;
  if (i < n) dst[i] = f2bf(src[i]);
}

__global__ void transpose_bf16_kernel(const float* __restrict__ W,
                                      unsigned short* __restrict__ Wt,
                                      int K, int N) {
  int i = blockIdx.x * blockDim.x + threadIdx.x;
  if (i < K * N) {
    int k = i / N, n = i % N;
    Wt[(size_t)n * K + k] = f2bf(W[i]);
  }
}

// ---------------------------------------------------------------------------
extern "C" void kernel_launch(void* const* d_in, const int* in_sizes, int n_in,
                              void* d_out, int out_size, void* d_ws, size_t ws_size,
                              hipStream_t stream) {
  (void)in_sizes; (void)n_in; (void)out_size; (void)ws_size;
  const int*   ids    = (const int*)d_in[0];
  const float* emb    = (const float*)d_in[1];
  const float* norm_g = (const float*)d_in[2];
  const float* norm_b = (const float*)d_in[3];
  const float* W1     = (const float*)d_in[4];
  const float* vc1    = (const float*)d_in[5];
  const float* b1     = (const float*)d_in[6];
  const float* lng1   = (const float*)d_in[7];
  const float* lnb1   = (const float*)d_in[8];
  const float* W2     = (const float*)d_in[9];
  const float* vc2    = (const float*)d_in[10];
  const float* b2     = (const float*)d_in[11];
  const float* lng2   = (const float*)d_in[12];
  const float* lnb2   = (const float*)d_in[13];
  const float* out_w  = (const float*)d_in[14];
  const float* out_b  = (const float*)d_in[15];
  float* out = (float*)d_out;

  constexpr int Bn = 1024, Ln = 64, E = 256, Dd = 128, Cc = 64;
  constexpr int M  = Ln * Bn;        // 65536 rows
  constexpr int N1 = 2 * Dd * 3;     // 768
  constexpr int K1 = E;              // 256 (same for layer 2: 2D = 256)

  char* ws = (char*)d_ws;
  float* X = (float*)ws;                     ws += (size_t)M * 256 * sizeof(float);
  float* H = (float*)ws;                     ws += (size_t)M * 256 * sizeof(float);
  unsigned short* XLN = (unsigned short*)ws; ws += (size_t)M * 256 * sizeof(unsigned short);
  float* U = (float*)ws;                     ws += (size_t)M * N1 * sizeof(float);
  unsigned short* W1t = (unsigned short*)ws; ws += (size_t)N1 * K1 * sizeof(unsigned short);
  unsigned short* W2t = (unsigned short*)ws; ws += (size_t)N1 * K1 * sizeof(unsigned short);
  unsigned short* OWt = (unsigned short*)ws; ws += (size_t)Cc * K1 * sizeof(unsigned short);

  // Weight prep: W (K,N) -> bf16 (N,K); out_w already (N,K) layout.
  transpose_bf16_kernel<<<(K1 * N1 + 255) / 256, 256, 0, stream>>>(W1, W1t, K1, N1);
  transpose_bf16_kernel<<<(K1 * N1 + 255) / 256, 256, 0, stream>>>(W2, W2t, K1, N1);
  convert_bf16_kernel<<<(Cc * K1 + 255) / 256, 256, 0, stream>>>(out_w, OWt, (size_t)Cc * K1);

  // Embedding gather + LN, transposed write to (L,B,E).
  embed_ln_kernel<<<M / 8, 256, 0, stream>>>(ids, emb, norm_g, norm_b, X, Bn, Ln);

  int gemm_blocks = ((M / 16) * (N1 / 64)) / 4;   // 4 waves per 128-thread block
  int scan_blocks = (Bn * 2 * Dd) / 256;

  // ---- SRU layer 1 ----
  ln_rows_bf16_kernel<<<M / 8, 256, 0, stream>>>(X, lng1, lnb1, XLN, M);
  gemm_bf16_wmma<<<gemm_blocks, 128, 0, stream>>>(XLN, W1t, U, nullptr, M, N1, K1, -1, 0);
  sru_scan_kernel<<<scan_blocks, 256, 0, stream>>>(U, X, vc1, b1, H, nullptr, Ln, Bn, Dd);

  // ---- SRU layer 2 ----
  ln_rows_bf16_kernel<<<M / 8, 256, 0, stream>>>(H, lng2, lnb2, XLN, M);
  gemm_bf16_wmma<<<gemm_blocks, 128, 0, stream>>>(XLN, W2t, U, nullptr, M, N1, K1, -1, 0);
  // Layer-2 scan writes h2 both as f32 (X) and fused bf16 (XLN) for the
  // output projection — removes a full 96 MB convert pass.
  sru_scan_kernel<<<scan_blocks, 256, 0, stream>>>(U, H, vc2, b2, X, XLN, Ln, Bn, Dd);

  // ---- Output projection: (L,B,256) @ out_w^T + out_b -> (B,L,64) ----
  int proj_blocks = ((M / 16) * (Cc / 64)) / 4;
  gemm_bf16_wmma<<<proj_blocks, 128, 0, stream>>>(XLN, OWt, out, out_b, M, Cc, K1, 10, 6);
}
